// MultiHeadAttention_58695023067794
// MI455X (gfx1250) — matrix operand order repaired and checked by reading, not compiled, not run on hardware
//
#include <hip/hip_runtime.h>

typedef __attribute__((ext_vector_type(16))) __bf16 v16bf;
typedef __attribute__((ext_vector_type(8)))  __bf16 v8bf;
typedef __attribute__((ext_vector_type(4)))  __bf16 v4bf;
typedef __attribute__((ext_vector_type(8)))  float  v8f;
typedef __attribute__((ext_vector_type(4)))  float  v4f;

constexpr int kB = 4, kS = 2048, kC = 1024, kH = 16, kD = 64, kHD = 1024;
constexpr int TM = 128, TN = 128, TK = 32;
constexpr int LDT = 40;   // padded LDS row stride (bf16 elems): 80 bytes, keeps 16B chunk alignment

static __device__ inline v16bf cat16(v8bf lo, v8bf hi) {
  return __builtin_shufflevector(lo, hi, 0,1,2,3,4,5,6,7,8,9,10,11,12,13,14,15);
}
static __device__ inline v8f wmma_bf16(v16bf a, v16bf b, v8f c) {
  // D = A(16x32) * B(32x16) + C, fp32 accumulate
  return __builtin_amdgcn_wmma_f32_16x16x32_bf16(false, a, false, b, (short)0, c, false, false);
}
// A-fragment (16x32, row-major source, stride ld): lane holds row (lane&15),
// chunks K=[kh*8,+8) and K=[16+kh*8,+8)
static __device__ inline v16bf frag_a_lds(const __bf16* base, int ld, int lane) {
  const int m = lane & 15, kh = lane >> 4;
  const __bf16* p = base + m * ld;
  v8bf lo = *(const v8bf*)(p + kh * 8);
  v8bf hi = *(const v8bf*)(p + 16 + kh * 8);
  return cat16(lo, hi);
}
// B-fragment (32x16) from [n][k]-major source: lane holds col (lane&15),
// chunks K=[kh*8,+8) and K=[16+kh*8,+8)
static __device__ inline v16bf frag_b_lds(const __bf16* base, int ld, int lane) {
  const int n = lane & 15, kh = lane >> 4;
  const __bf16* p = base + n * ld;
  v8bf lo = *(const v8bf*)(p + kh * 8);
  v8bf hi = *(const v8bf*)(p + 16 + kh * 8);
  return cat16(lo, hi);
}
static __device__ inline v8f vzero8() {
  v8f z = {0.f,0.f,0.f,0.f,0.f,0.f,0.f,0.f};
  return z;
}

// stage 128x32 fp32 tile (row-major, row stride rs) -> bf16 LDS, packed 8B stores
static __device__ inline void stage_a_f32(__bf16* dst, const float* src, size_t rs,
                                          int m0, int k0, int tid) {
  #pragma unroll
  for (int it = 0; it < 4; ++it) {
    int id = tid + it * 256;                 // 0..1023
    int m  = id >> 3;
    int k4 = (id & 7) * 4;
    v4f f = *(const v4f*)(src + (size_t)(m0 + m) * rs + k0 + k4);
    v4bf o;
    o[0] = (__bf16)f[0]; o[1] = (__bf16)f[1]; o[2] = (__bf16)f[2]; o[3] = (__bf16)f[3];
    *(v4bf*)&dst[m * LDT + k4] = o;
  }
}

// shared WMMA tile compute: 2 row-tiles x 4 col-tiles per wave
static __device__ inline void mma_tile(const __bf16* AsBuf, const __bf16* BsBuf,
                                       int wr, int wc, int lane, v8f (&acc)[2][4]) {
  v16bf a0 = frag_a_lds(&AsBuf[(wr * 32 +  0) * LDT], LDT, lane);
  v16bf a1 = frag_a_lds(&AsBuf[(wr * 32 + 16) * LDT], LDT, lane);
  #pragma unroll
  for (int ct = 0; ct < 4; ++ct) {
    v16bf b = frag_b_lds(&BsBuf[(wc * 64 + ct * 16) * LDT], LDT, lane);
    acc[0][ct] = wmma_bf16(a0, b, acc[0][ct]);
    acc[1][ct] = wmma_bf16(a1, b, acc[1][ct]);
  }
}

// ---------------- Kernel 1: fused QKV projection ----------------
// [B*S, C] x [C, 3*H*D]  (fp32 -> bf16), scatter q/k [bh][s][d], vT [bh][d][s]
__global__ __launch_bounds__(256) void qkv_proj_kernel(
    const float* __restrict__ x,
    const float* __restrict__ Wq, const float* __restrict__ Wk, const float* __restrict__ Wv,
    __bf16* __restrict__ qws, __bf16* __restrict__ kws, __bf16* __restrict__ vTws)
{
  __shared__ __bf16 As[2][TM * LDT];
  __shared__ __bf16 Bs[2][TN * LDT];
  const int tid  = threadIdx.x;
  const int lane = tid & 31;
  const int wave = tid >> 5;
  const int m0 = blockIdx.x * TM;            // rows over B*S
  const int n0 = blockIdx.y * TN;            // cols over 3*H*D = 3072
  const int which = n0 >> 10;                // 0=q 1=k 2=v (tiles never straddle)
  const float* W = (which == 0) ? Wq : (which == 1) ? Wk : Wv;

  const int wr = wave >> 1;                  // rows wr*32..+31
  const int wc = wave & 1;                   // cols wc*64..+63

  v8f acc[2][4];
  #pragma unroll
  for (int i = 0; i < 2; ++i)
    #pragma unroll
    for (int j = 0; j < 4; ++j) acc[i][j] = vzero8();

  // B staging: Bs[n][k] = W[h][c=k0+k][d] (transpose scatter)
  auto stage_b = [&](__bf16* dst, int k0) {
    #pragma unroll
    for (int it = 0; it < 4; ++it) {
      int id = tid + it * 256;               // 0..1023
      int c  = id >> 5;                      // 0..31
      int n4 = (id & 31) * 4;                // 0..124
      int ng = (n0 & 1023) + n4;
      int h  = ng >> 6;
      int d  = ng & 63;
      v4f f = *(const v4f*)(W + (size_t)h * kC * kD + (size_t)(k0 + c) * kD + d);
      dst[(n4 + 0) * LDT + c] = (__bf16)f[0];
      dst[(n4 + 1) * LDT + c] = (__bf16)f[1];
      dst[(n4 + 2) * LDT + c] = (__bf16)f[2];
      dst[(n4 + 3) * LDT + c] = (__bf16)f[3];
    }
  };

  constexpr int NK = kC / TK;
  stage_a_f32(As[0], x, kC, m0, 0, tid);
  stage_b(Bs[0], 0);
  __syncthreads();
  int cur = 0;
  for (int kb = 0; kb < NK; ++kb) {
    const int nxt = cur ^ 1;
    if (kb + 1 < NK) {
      stage_a_f32(As[nxt], x, kC, m0, (kb + 1) * TK, tid);
      stage_b(Bs[nxt], (kb + 1) * TK);
    }
    mma_tile(As[cur], Bs[cur], wr, wc, lane, acc);
    __syncthreads();
    cur = nxt;
  }

  // uniform-branch epilogue, linearized addressing
  const int lh = lane >> 4, nl = lane & 15;
  const int mbase = m0 + wr * 32 + lh * 8;
  if (which < 2) {
    __bf16* __restrict__ dst = (which == 0) ? qws : kws;
    #pragma unroll
    for (int ct = 0; ct < 4; ++ct) {
      const int ng = ((n0 + wc * 64 + ct * 16) & 1023) + nl;
      const int h = ng >> 6, d = ng & 63;
      #pragma unroll
      for (int rt = 0; rt < 2; ++rt) {
        const int mg = mbase + rt * 16;
        const int bb = mg >> 11;
        const int s0 = mg & 2047;
        __bf16* p = dst + ((size_t)(bb * kH + h) * kS + s0) * kD + d;
        #pragma unroll
        for (int e = 0; e < 8; ++e)
          p[(size_t)e * kD] = (__bf16)acc[rt][ct][e];
      }
    }
  } else {
    #pragma unroll
    for (int ct = 0; ct < 4; ++ct) {
      const int ng = ((n0 + wc * 64 + ct * 16) & 1023) + nl;
      const int h = ng >> 6, d = ng & 63;
      #pragma unroll
      for (int rt = 0; rt < 2; ++rt) {
        const int mg = mbase + rt * 16;
        const int bb = mg >> 11;
        const int s0 = mg & 2047;
        // vT[bh][d][s]: lane's 8 elements are contiguous in s -> one 16B store
        v8bf o;
        #pragma unroll
        for (int e = 0; e < 8; ++e) o[e] = (__bf16)acc[rt][ct][e];
        *(v8bf*)(vTws + ((size_t)(bb * kH + h) * kD + d) * kS + s0) = o;
      }
    }
  }
}

// ---------------- Kernel 2: causal flash attention ----------------
// block = 8 waves = 8 consecutive 16-row q tiles of one (b,h).
// K/V tiles staged to LDS once per block (8x reuse), double-buffered.
__global__ __launch_bounds__(256) void attn_kernel(
    const __bf16* __restrict__ qws, const __bf16* __restrict__ kws,
    const __bf16* __restrict__ vTws, __bf16* __restrict__ attn)
{
  constexpr int LDP = 40, LDK = 72, LDV = 40;
  __shared__ __bf16 Ks[2][32 * LDK];         // [key][d]  (B-frag layout for QK^T)
  __shared__ __bf16 Vs[2][64 * LDV];         // [d][t]    (B-frag layout for P*V)
  __shared__ __bf16 Plds[8 * 16 * LDP];      // per-wave 16x32 P tile
  const int tid = threadIdx.x, lane = tid & 31, wave = tid >> 5;
  const int bh  = blockIdx.x >> 4;           // 16 blocks per (b,h)
  const int qt0 = (blockIdx.x & 15) * 8;     // first q-tile of this block
  const int qbase = (qt0 + wave) * 16;       // this wave's q rows
  const int bb = bh / kH, h = bh % kH;
  const int lh = lane >> 4, nl = lane & 15;

  const __bf16* qp = qws + (size_t)bh * kS * kD;
  const __bf16* kp = kws + (size_t)bh * kS * kD;
  const __bf16* vp = vTws + (size_t)bh * kD * kS;

  // cooperative staging of one 32-key step: K 32x64, V 64x32 (bf16, 16B ops)
  auto stage_kv = [&](int buf, int j0) {
    {
      int key = tid >> 3, d8 = (tid & 7) * 8;
      *(v8bf*)&Ks[buf][key * LDK + d8] =
          *(const v8bf*)(kp + (size_t)(j0 + key) * kD + d8);
    }
    {
      int d = tid >> 2, t8 = (tid & 3) * 8;
      *(v8bf*)&Vs[buf][d * LDV + t8] =
          *(const v8bf*)(vp + (size_t)d * kS + j0 + t8);
    }
  };

  // Q fragments: rows qbase..+15, K-dim = head dim split 2x32
  v16bf Qa[2];
  {
    const __bf16* qrow = qp + (size_t)(qbase + nl) * kD;
    #pragma unroll
    for (int f = 0; f < 2; ++f) {
      v8bf lo = *(const v8bf*)(qrow + f * 32 + lh * 8);
      v8bf hi = *(const v8bf*)(qrow + f * 32 + 16 + lh * 8);
      Qa[f] = cat16(lo, hi);
    }
  }

  v8f O[4];
  #pragma unroll
  for (int t = 0; t < 4; ++t) O[t] = vzero8();
  float mrow[8], lrow[8];
  #pragma unroll
  for (int e = 0; e < 8; ++e) { mrow[e] = -__builtin_inff(); lrow[e] = 0.f; }

  __bf16* P = &Plds[wave * 16 * LDP];
  const int nsteps = (qt0 * 16 + 112 + 16 + 31) >> 5;  // causal bound of block's last tile

  stage_kv(0, 0);
  __syncthreads();
  int cur = 0;
  for (int kb = 0; kb < nsteps; ++kb) {
    const int j0 = kb * 32;
    const int nxt = cur ^ 1;
    if (kb + 1 < nsteps) stage_kv(nxt, (kb + 1) * 32);

    if (j0 <= qbase + 15) {                  // wave-uniform: skip fully-masked steps
      v8f Sc[2] = { vzero8(), vzero8() };
      #pragma unroll
      for (int jb = 0; jb < 2; ++jb)
        #pragma unroll
        for (int f = 0; f < 2; ++f) {
          v16bf Kb = frag_b_lds(&Ks[cur][(jb * 16) * LDK + f * 32], LDK, lane);
          Sc[jb] = wmma_bf16(Qa[f], Kb, Sc[jb]);
        }
      // scale + causal mask + online softmax (row = lh*8+e, col = jb*16+nl)
      float tmax[8], pv[2][8], alpha[8];
      #pragma unroll
      for (int e = 0; e < 8; ++e) {
        const int i = qbase + lh * 8 + e;
        #pragma unroll
        for (int jb = 0; jb < 2; ++jb) {
          const int j = j0 + jb * 16 + nl;
          float s = Sc[jb][e] * 0.125f;      // 1/sqrt(64)
          Sc[jb][e] = (j <= i) ? s : -__builtin_inff();
        }
        tmax[e] = fmaxf(Sc[0][e], Sc[1][e]);
      }
      #pragma unroll
      for (int mask = 1; mask < 16; mask <<= 1)
        #pragma unroll
        for (int e = 0; e < 8; ++e)
          tmax[e] = fmaxf(tmax[e], __shfl_xor(tmax[e], mask, 32));
      #pragma unroll
      for (int e = 0; e < 8; ++e) {
        float mn = fmaxf(mrow[e], tmax[e]);
        alpha[e] = __expf(mrow[e] - mn);
        mrow[e] = mn;
        float p0 = __expf(Sc[0][e] - mn);
        float p1 = __expf(Sc[1][e] - mn);
        pv[0][e] = p0; pv[1][e] = p1;
        Sc[0][e] = p0 + p1;                  // reuse for row-sum reduce
      }
      #pragma unroll
      for (int mask = 1; mask < 16; mask <<= 1)
        #pragma unroll
        for (int e = 0; e < 8; ++e)
          Sc[0][e] += __shfl_xor(Sc[0][e], mask, 32);
      #pragma unroll
      for (int e = 0; e < 8; ++e) lrow[e] = lrow[e] * alpha[e] + Sc[0][e];
      #pragma unroll
      for (int t = 0; t < 4; ++t)
        #pragma unroll
        for (int e = 0; e < 8; ++e) O[t][e] *= alpha[e];

      // re-layout P (C-layout regs -> A-fragment) through per-wave LDS tile
      #pragma unroll
      for (int jb = 0; jb < 2; ++jb)
        #pragma unroll
        for (int e = 0; e < 8; ++e)
          P[(lh * 8 + e) * LDP + jb * 16 + nl] = (__bf16)pv[jb][e];
      asm volatile("s_wait_dscnt 0x0" ::: "memory");
      v16bf Pa = frag_a_lds(P, LDP, lane);

      // O += P(16x32) * V(32x64)
      #pragma unroll
      for (int nb = 0; nb < 4; ++nb) {
        v16bf Vb = frag_b_lds(&Vs[cur][(nb * 16) * LDV], LDV, lane);
        O[nb] = wmma_bf16(Pa, Vb, O[nb]);
      }
    }
    __syncthreads();
    cur = nxt;
  }

  float linv[8];
  #pragma unroll
  for (int e = 0; e < 8; ++e) linv[e] = 1.0f / lrow[e];
  #pragma unroll
  for (int nb = 0; nb < 4; ++nb) {
    __bf16* p = attn + ((size_t)bb * kS + qbase + lh * 8) * kHD + h * kD + nb * 16 + nl;
    #pragma unroll
    for (int e = 0; e < 8; ++e)
      p[(size_t)e * kHD] = (__bf16)(O[nb][e] * linv[e]);
  }
}

// ---------------- Kernel 3: output projection + bias ----------------
// [B*S, HD](bf16) x Wp[HD, C](fp32->bf16) + bp -> out fp32
__global__ __launch_bounds__(256) void out_proj_kernel(
    const __bf16* __restrict__ attn, const float* __restrict__ Wp,
    const float* __restrict__ bp, float* __restrict__ out)
{
  __shared__ __bf16 As[2][TM * LDT];
  __shared__ __bf16 Bs[2][TN * LDT];
  const int tid  = threadIdx.x;
  const int lane = tid & 31;
  const int wave = tid >> 5;
  const int m0 = blockIdx.x * TM;
  const int n0 = blockIdx.y * TN;
  const int wr = wave >> 1;
  const int wc = wave & 1;

  v8f acc[2][4];
  #pragma unroll
  for (int i = 0; i < 2; ++i)
    #pragma unroll
    for (int j = 0; j < 4; ++j) acc[i][j] = vzero8();

  auto stage_a = [&](__bf16* dst, int k0) {    // attn already bf16, 16B vector copies
    #pragma unroll
    for (int it = 0; it < 2; ++it) {
      int id = tid + it * 256;                 // 0..511
      int m  = id >> 2;
      int k8 = (id & 3) * 8;
      v8bf v = *(const v8bf*)(attn + (size_t)(m0 + m) * kHD + k0 + k8);
      *(v8bf*)&dst[m * LDT + k8] = v;
    }
  };
  auto stage_b = [&](__bf16* dst, int k0) {    // Bs[n][k] = Wp[k0+k][n0+n]
    #pragma unroll
    for (int it = 0; it < 4; ++it) {
      int id = tid + it * 256;
      int c  = id >> 5;
      int n4 = (id & 31) * 4;
      v4f f = *(const v4f*)(Wp + (size_t)(k0 + c) * kC + n0 + n4);
      dst[(n4 + 0) * LDT + c] = (__bf16)f[0];
      dst[(n4 + 1) * LDT + c] = (__bf16)f[1];
      dst[(n4 + 2) * LDT + c] = (__bf16)f[2];
      dst[(n4 + 3) * LDT + c] = (__bf16)f[3];
    }
  };

  constexpr int NK = kHD / TK;
  stage_a(As[0], 0);
  stage_b(Bs[0], 0);
  __syncthreads();
  int cur = 0;
  for (int kb = 0; kb < NK; ++kb) {
    const int nxt = cur ^ 1;
    if (kb + 1 < NK) {
      stage_a(As[nxt], (kb + 1) * TK);
      stage_b(Bs[nxt], (kb + 1) * TK);
    }
    mma_tile(As[cur], Bs[cur], wr, wc, lane, acc);
    __syncthreads();
    cur = nxt;
  }

  const int lh = lane >> 4, nl = lane & 15;
  const int mbase = m0 + wr * 32 + lh * 8;
  #pragma unroll
  for (int ct = 0; ct < 4; ++ct) {
    const int ng = n0 + wc * 64 + ct * 16 + nl;
    const float bias = bp[ng];
    #pragma unroll
    for (int rt = 0; rt < 2; ++rt) {
      float* p = out + (size_t)(mbase + rt * 16) * kC + ng;
      #pragma unroll
      for (int e = 0; e < 8; ++e)
        p[(size_t)e * kC] = acc[rt][ct][e] + bias;
    }
  }
}

extern "C" void kernel_launch(void* const* d_in, const int* in_sizes, int n_in,
                              void* d_out, int out_size, void* d_ws, size_t ws_size,
                              hipStream_t stream) {
  const float* x  = (const float*)d_in[0];
  const float* Wq = (const float*)d_in[1];
  const float* Wk = (const float*)d_in[2];
  const float* Wv = (const float*)d_in[3];
  const float* Wp = (const float*)d_in[4];
  const float* bp = (const float*)d_in[5];
  float* out = (float*)d_out;

  const size_t qkN = (size_t)kB * kH * kS * kD;   // 8,388,608 elems per tensor
  __bf16* qws  = (__bf16*)d_ws;
  __bf16* kws  = qws + qkN;
  __bf16* vT   = kws + qkN;
  __bf16* attn = vT + qkN;                        // total 64 MB bf16 scratch

  qkv_proj_kernel<<<dim3((kB * kS) / TM, (3 * kHD) / TN), 256, 0, stream>>>(
      x, Wq, Wk, Wv, qws, kws, vT);
  attn_kernel<<<(kB * kH * (kS / 16)) / 8, 256, 0, stream>>>(qws, kws, vT, attn);
  out_proj_kernel<<<dim3((kB * kS) / TM, kC / TN), 256, 0, stream>>>(attn, Wp, bp, out);
}
